// SelfAttention_30837865185734
// MI455X (gfx1250) — matrix operand order repaired and checked by reading, not compiled, not run on hardware
//
#include <hip/hip_runtime.h>
#include <math.h>

#define DIM 1024
#define SEQ 4096

typedef __attribute__((ext_vector_type(16))) __bf16        v16bf;
typedef __attribute__((ext_vector_type(8)))  float         v8f;
typedef __attribute__((ext_vector_type(4)))  unsigned int  u32x4;

union Frag { u32x4 u[2]; v16bf v; };

// ---------------------------------------------------------------------------
// Fragment loaders per CDNA5 §7.12.2 VGPR layouts (16-bit A 16x32, B 32x16).
// ---------------------------------------------------------------------------

// A (16x32 bf16) from row-major [*, ld]: lane<16 -> K {kk..kk+7},{kk+16..kk+23}
//                                        lane>=16 -> K {kk+8..},{kk+24..}
__device__ __forceinline__ v16bf load_a_frag(const __bf16* base, int row, int ld,
                                             int kk, int lane) {
    const __bf16* p = base + (size_t)row * ld + kk + ((lane >> 4) << 3);
    Frag f;
    f.u[0] = *(const u32x4*)(p);        // 8 halves: K lo group
    f.u[1] = *(const u32x4*)(p + 16);   // 8 halves: K hi group
    return f.v;
}

// B (32x16 bf16) from TRANSPOSED row-major source Bt[N][Kdim]:
// lane<16 -> col=lane, K {kk..kk+7},{kk+16..kk+23}; lane>=16 -> col=lane-16, K {kk+8..},{kk+24..}
__device__ __forceinline__ v16bf load_b_frag(const __bf16* baseT, int col, int ld,
                                             int kk, int lane) {
    const __bf16* p = baseT + (size_t)col * ld + kk + ((lane >> 4) << 3);
    Frag f;
    f.u[0] = *(const u32x4*)(p);
    f.u[1] = *(const u32x4*)(p + 16);
    return f.v;
}

// ---------------------------------------------------------------------------
// Generic WMMA GEMM:  Out = alpha * (A @ Bt^T) + bias
//   A  : bf16 row-major [M, K]
//   Bt : bf16 row-major [N, K]   (the B operand, pre-transposed)
// OUT_MODE: 0 = f32 row-major, 1 = bf16 row-major, 2 = bf16 transposed store
// Each wave computes a 32x64 output strip: 2 (M) x 4 (N) WMMA tiles,
// 8 accumulators; per K-step: 12 x b128 loads feed 8 v_wmma ops.
// ---------------------------------------------------------------------------
template <bool ADD_BIAS, int OUT_MODE>
__global__ __launch_bounds__(256) void gemm_wmma(const __bf16* __restrict__ A,
                                                 const __bf16* __restrict__ Bt,
                                                 const float* __restrict__ bias,
                                                 void* __restrict__ Out,
                                                 int M, int N, int K,
                                                 float alpha, int out_ld) {
    const int lane = threadIdx.x & 31;
    const int wave = threadIdx.x >> 5;
    const int n4Tiles = N >> 6;                       // strips of 64 cols
    const int work = blockIdx.x * 8 + wave;           // 8 waves per block
    const int mi = work / n4Tiles;                    // 32-row tile index
    const int nj = work % n4Tiles;                    // 64-col strip index
    if (mi >= (M >> 5)) return;

    const int n0 = lane & 15;
    const int aRow0 = mi * 32 + n0;                   // A rows for this lane
    const int aRow1 = mi * 32 + 16 + n0;

    v8f acc[2][4] = {};

    for (int kk = 0; kk < K; kk += 32) {
        v16bf a0 = load_a_frag(A, aRow0, K, kk, lane);
        v16bf a1 = load_a_frag(A, aRow1, K, kk, lane);
        if (kk + 32 < K) {                            // speculative L2 prefetch
            __builtin_prefetch(A + (size_t)aRow0 * K + kk + 32, 0, 1);
            __builtin_prefetch(A + (size_t)aRow1 * K + kk + 32, 0, 1);
        }

        v16bf b0 = load_b_frag(Bt, nj * 64 +  0 + n0, K, kk, lane);
        v16bf b1 = load_b_frag(Bt, nj * 64 + 16 + n0, K, kk, lane);
        v16bf b2 = load_b_frag(Bt, nj * 64 + 32 + n0, K, kk, lane);
        v16bf b3 = load_b_frag(Bt, nj * 64 + 48 + n0, K, kk, lane);

        acc[0][0] = __builtin_amdgcn_wmma_f32_16x16x32_bf16(false, a0, false, b0,
                                                            (short)0, acc[0][0], false, false);
        acc[0][1] = __builtin_amdgcn_wmma_f32_16x16x32_bf16(false, a0, false, b1,
                                                            (short)0, acc[0][1], false, false);
        acc[0][2] = __builtin_amdgcn_wmma_f32_16x16x32_bf16(false, a0, false, b2,
                                                            (short)0, acc[0][2], false, false);
        acc[0][3] = __builtin_amdgcn_wmma_f32_16x16x32_bf16(false, a0, false, b3,
                                                            (short)0, acc[0][3], false, false);
        acc[1][0] = __builtin_amdgcn_wmma_f32_16x16x32_bf16(false, a1, false, b0,
                                                            (short)0, acc[1][0], false, false);
        acc[1][1] = __builtin_amdgcn_wmma_f32_16x16x32_bf16(false, a1, false, b1,
                                                            (short)0, acc[1][1], false, false);
        acc[1][2] = __builtin_amdgcn_wmma_f32_16x16x32_bf16(false, a1, false, b2,
                                                            (short)0, acc[1][2], false, false);
        acc[1][3] = __builtin_amdgcn_wmma_f32_16x16x32_bf16(false, a1, false, b3,
                                                            (short)0, acc[1][3], false, false);
    }

    // Epilogue. C/D layout: lane<16 -> N=lane, M=vgpr i; lane>=16 -> M=i+8.
    const int mhi = (lane >> 4) * 8;
#pragma unroll
    for (int r = 0; r < 2; ++r) {
#pragma unroll
        for (int t = 0; t < 4; ++t) {
            const int n = nj * 64 + t * 16 + n0;
            const float badd = ADD_BIAS ? bias[n] : 0.0f;
#pragma unroll
            for (int i = 0; i < 8; ++i) {
                const float val = acc[r][t][i] * alpha + badd;
                const int m = mi * 32 + r * 16 + mhi + i;
                if (OUT_MODE == 0) {
                    ((float*)Out)[(size_t)m * out_ld + n] = val;
                } else if (OUT_MODE == 1) {
                    ((__bf16*)Out)[(size_t)m * out_ld + n] = (__bf16)val;
                } else {
                    ((__bf16*)Out)[(size_t)n * out_ld + m] = (__bf16)val;  // transposed
                }
            }
        }
    }
}

// ---------------------------------------------------------------------------
// fp32 -> bf16 conversion (row-major copy)
// ---------------------------------------------------------------------------
__global__ __launch_bounds__(256) void cvt_bf16(const float* __restrict__ in,
                                                __bf16* __restrict__ out, int n) {
    int i = blockIdx.x * blockDim.x + threadIdx.x;
    if (i < n) out[i] = (__bf16)in[i];
}

// fp32 [rows,cols] -> bf16 transposed [cols,rows]
__global__ __launch_bounds__(256) void cvt_bf16_t(const float* __restrict__ in,
                                                  __bf16* __restrict__ out,
                                                  int rows, int cols) {
    int i = blockIdx.x * blockDim.x + threadIdx.x;
    if (i < rows * cols) {
        int r = i / cols, c = i % cols;
        out[(size_t)c * rows + r] = (__bf16)in[i];
    }
}

// ---------------------------------------------------------------------------
// Row softmax: S (f32 [SEQ]) -> P (bf16 [SEQ]); one 256-thread block per row.
// ---------------------------------------------------------------------------
__global__ __launch_bounds__(256) void softmax_rows(const float* __restrict__ S,
                                                    __bf16* __restrict__ P, int n) {
    const int row = blockIdx.x;
    const float* x = S + (size_t)row * n;
    __bf16* p = P + (size_t)row * n;
    __shared__ float red[256];

    float lmax = -INFINITY;
    for (int i = threadIdx.x; i < n; i += 256) lmax = fmaxf(lmax, x[i]);
    red[threadIdx.x] = lmax;
    __syncthreads();
    for (int s = 128; s > 0; s >>= 1) {
        if (threadIdx.x < s)
            red[threadIdx.x] = fmaxf(red[threadIdx.x], red[threadIdx.x + s]);
        __syncthreads();
    }
    const float m = red[0];
    __syncthreads();

    float lsum = 0.0f;
    for (int i = threadIdx.x; i < n; i += 256) lsum += __expf(x[i] - m);
    red[threadIdx.x] = lsum;
    __syncthreads();
    for (int s = 128; s > 0; s >>= 1) {
        if (threadIdx.x < s) red[threadIdx.x] += red[threadIdx.x + s];
        __syncthreads();
    }
    const float inv = 1.0f / red[0];

    for (int i = threadIdx.x; i < n; i += 256)
        p[i] = (__bf16)(__expf(x[i] - m) * inv);
}

// ---------------------------------------------------------------------------
// Launch: X->Q,K,V (bf16 WMMA + bias), S = scale*Q@K^T (f32), softmax -> P
// (bf16), Out = P@V (f32). Workspace layout (needs ~134 MB of d_ws):
//   Xb 8M | Wqt 2M | Wkt 2M | Wvt 2M | Qb 8M | Kb 8M | Vt 8M | S 64M | P 32M
// ---------------------------------------------------------------------------
extern "C" void kernel_launch(void* const* d_in, const int* in_sizes, int n_in,
                              void* d_out, int out_size, void* d_ws, size_t ws_size,
                              hipStream_t stream) {
    const float* X  = (const float*)d_in[0];
    const float* Wq = (const float*)d_in[1];
    const float* bq = (const float*)d_in[2];
    const float* Wk = (const float*)d_in[3];
    const float* bk = (const float*)d_in[4];
    const float* Wv = (const float*)d_in[5];
    const float* bv = (const float*)d_in[6];
    float* out = (float*)d_out;

    char* ws = (char*)d_ws;
    size_t off = 0;
    __bf16* Xb  = (__bf16*)(ws + off); off += (size_t)SEQ * DIM * 2;
    __bf16* Wqt = (__bf16*)(ws + off); off += (size_t)DIM * DIM * 2;
    __bf16* Wkt = (__bf16*)(ws + off); off += (size_t)DIM * DIM * 2;
    __bf16* Wvt = (__bf16*)(ws + off); off += (size_t)DIM * DIM * 2;
    __bf16* Qb  = (__bf16*)(ws + off); off += (size_t)SEQ * DIM * 2;
    __bf16* Kb  = (__bf16*)(ws + off); off += (size_t)SEQ * DIM * 2;
    __bf16* Vt  = (__bf16*)(ws + off); off += (size_t)DIM * SEQ * 2;
    float*  Sc  = (float*)(ws + off);  off += (size_t)SEQ * SEQ * 4;
    __bf16* P   = (__bf16*)(ws + off); off += (size_t)SEQ * SEQ * 2;

    // 1) Convert inputs to bf16 (W transposed so WMMA B-fragments load contiguously)
    cvt_bf16<<<(SEQ * DIM) / 256, 256, 0, stream>>>(X, Xb, SEQ * DIM);
    cvt_bf16_t<<<(DIM * DIM) / 256, 256, 0, stream>>>(Wq, Wqt, DIM, DIM);
    cvt_bf16_t<<<(DIM * DIM) / 256, 256, 0, stream>>>(Wk, Wkt, DIM, DIM);
    cvt_bf16_t<<<(DIM * DIM) / 256, 256, 0, stream>>>(Wv, Wvt, DIM, DIM);

    // 2) QKV projections. work = (SEQ/32)*(DIM/64) = 2048 waves -> 256 blocks.
    gemm_wmma<true, 1><<<256, 256, 0, stream>>>(Xb, Wqt, bq, Qb,
                                                SEQ, DIM, DIM, 1.0f, DIM);
    gemm_wmma<true, 1><<<256, 256, 0, stream>>>(Xb, Wkt, bk, Kb,
                                                SEQ, DIM, DIM, 1.0f, DIM);
    gemm_wmma<true, 2><<<256, 256, 0, stream>>>(Xb, Wvt, bv, Vt,
                                                SEQ, DIM, DIM, 1.0f, SEQ);  // V stored [D,S]

    // 3) Scores = (1/sqrt(D)) * Q @ K^T  (B operand transposed source == K row-major)
    //    work = (SEQ/32)*(SEQ/64) = 8192 waves -> 1024 blocks.
    gemm_wmma<false, 0><<<1024, 256, 0, stream>>>(Qb, Kb, nullptr, Sc,
                                                  SEQ, SEQ, DIM, 0.03125f, SEQ);

    // 4) Row softmax -> bf16 probabilities
    softmax_rows<<<SEQ, 256, 0, stream>>>(Sc, P, SEQ);

    // 5) Out = P @ V  (B operand transposed source == Vt[D,S]); f32 to d_out.
    gemm_wmma<false, 0><<<256, 256, 0, stream>>>(P, Vt, nullptr, out,
                                                 SEQ, DIM, SEQ, 1.0f, DIM);
}
